// Annotation_model_71683004170723
// MI455X (gfx1250) — matrix operand order repaired and checked by reading, not compiled, not run on hardware
//
#include <hip/hip_runtime.h>

#define N_NODES 50000
#define N_EDGES 640000
#define D 128
#define N_CLS 64
#define NT 5               // node tiles (of 16 rows) per block
#define NBLK (N_NODES / (16 * NT))   // 625
#define WPITCH 136         // LDS row pitch in halves: 128 + 8 (16B pad -> conflict-free)

typedef __attribute__((ext_vector_type(16))) _Float16 v16h;
typedef __attribute__((ext_vector_type(8)))  _Float16 v8h;
typedef __attribute__((ext_vector_type(4)))  _Float16 v4h;
typedef __attribute__((ext_vector_type(8)))  float    v8f;
typedef unsigned u32x4 __attribute__((ext_vector_type(4)));
typedef int      i32x4 __attribute__((ext_vector_type(4)));
typedef int      i32x8 __attribute__((ext_vector_type(8)));

#if defined(__has_builtin)
#if __has_builtin(__builtin_amdgcn_tensor_load_to_lds) && \
    __has_builtin(__builtin_amdgcn_s_wait_tensorcnt)
#define HAVE_TDM 1
#endif
#endif
#ifndef HAVE_TDM
#define HAVE_TDM 0
#endif

// ---------------------------------------------------------------------------
// WMMA tile loads (wave32, v_wmma_f32_16x16x32_f16)
// A (16x32 MxK):  lane<16: M=lane,   v[0..7]=K(klo..+7), v[8..15]=K(klo+16..+23), klo=kt*32
//                 lane>=16: M=lane-16, klo=kt*32+8
// B (32x16 KxN):  lane<16: N=lane,   v[0..7]=K(kb..+7), v[8..15]=K(kb+16..+23), kb=kt*32
//                 lane>=16: N=lane-16, kb=kt*32+8
// ---------------------------------------------------------------------------
__device__ __forceinline__ v16h ldA_f16(const _Float16* __restrict__ rowp, int klo) {
  v8h lo = *(const v8h*)(rowp + klo);
  v8h hi = *(const v8h*)(rowp + klo + 16);
  v16h v;
#pragma unroll
  for (int i = 0; i < 8; ++i) { v[i] = lo[i]; v[8 + i] = hi[i]; }
  return v;
}

__device__ __forceinline__ v16h ldB_f16(const _Float16* __restrict__ rowp, int kb) {
  v8h lo = *(const v8h*)(rowp + kb);
  v8h hi = *(const v8h*)(rowp + kb + 16);
  v16h v;
#pragma unroll
  for (int i = 0; i < 8; ++i) { v[i] = lo[i]; v[8 + i] = hi[i]; }
  return v;
}

// ---------------------------------------------------------------------------
// TDM: DMA a rows x 128 f16 tile (row stride 128) global -> LDS with a
// 4-DWORD pad every 64 DWORDs (one 256B row) => LDS pitch 136 halves.
// D# built per ISA 8.3/8.4 (group0: count/lds/global/type, group1: dims).
// ---------------------------------------------------------------------------
__device__ __forceinline__ unsigned lds_off(const void* p) {
  return (unsigned)(unsigned long long)(uintptr_t)p;  // low 32 bits = LDS byte addr
}

#if HAVE_TDM
__device__ __forceinline__ void tdm_load_f16(unsigned ldsoff, const _Float16* g, int rows) {
  unsigned long long ga = (unsigned long long)(uintptr_t)g;
  u32x4 g0;
  g0[0] = 1u;                                            // count=1
  g0[1] = ldsoff;                                        // lds_addr
  g0[2] = (unsigned)ga;                                  // global_addr[31:0]
  g0[3] = (unsigned)((ga >> 32) & 0x1FFFFFFu) | (2u << 30);  // addr[56:32] | type=2
  i32x8 g1;
  g1[0] = (1 << 16) | (1 << 20) | (5 << 22) | (3 << 25); // 2B elems, pad: 64DW interval, 4DW amount
  g1[1] = (D & 0xffff) << 16;                            // tensor_dim0 lo
  g1[2] = (rows & 0xffff) << 16;                         // dim0 hi=0 | tensor_dim1 lo
  g1[3] = (D & 0xffff) << 16;                            // dim1 hi=0 | tile_dim0=128
  g1[4] = rows & 0xffff;                                 // tile_dim1=rows, tile_dim2=0
  g1[5] = D;                                             // tensor_dim0_stride
  g1[6] = 0;
  g1[7] = 0;
  i32x4 z4 = {0, 0, 0, 0};
#if __clang_major__ >= 23
  i32x8 z8 = {0, 0, 0, 0, 0, 0, 0, 0};
  __builtin_amdgcn_tensor_load_to_lds(g0, g1, z4, z4, z8, 0);
#else
  __builtin_amdgcn_tensor_load_to_lds(g0, g1, z4, z4, 0);
#endif
}
#else
__device__ __forceinline__ void stage_w_fallback(_Float16* dst, const _Float16* src, int rows) {
  const int chunks = rows * (D / 8);  // 16B chunks per row
  for (int i = threadIdx.x; i < chunks; i += blockDim.x) {
    const int rr = i >> 4, cc = i & 15;
    *(uint4*)(dst + rr * WPITCH + cc * 8) = *(const uint4*)(src + rr * D + cc * 8);
  }
}
#endif

// ---------------------------------------------------------------------------
// f32 -> f16 pre-conversion (weights + x), 4 elems/thread
// ---------------------------------------------------------------------------
__global__ __launch_bounds__(256) void cvt4_kernel(const float* __restrict__ in,
                                                   _Float16* __restrict__ out, int n4) {
  const int i = blockIdx.x * blockDim.x + threadIdx.x;
  if (i >= n4) return;
  float4 f = ((const float4*)in)[i];
  v4h h;
  h[0] = (_Float16)f.x; h[1] = (_Float16)f.y;
  h[2] = (_Float16)f.z; h[3] = (_Float16)f.w;
  ((v4h*)out)[i] = h;
}

// mean = agg / max(deg,1), converted to f16.  Row = 32 float4 groups.
__global__ __launch_bounds__(256) void mean_cvt_kernel(
    const float* __restrict__ agg, const float* __restrict__ deg,
    _Float16* __restrict__ outh, int n4) {
  const int i = blockIdx.x * blockDim.x + threadIdx.x;
  if (i >= n4) return;
  const int node = i >> 5;  // (D/4)=32 groups per row
  const float sc = 1.0f / fmaxf(deg[node], 1.0f);
  float4 f = ((const float4*)agg)[i];
  v4h h;
  h[0] = (_Float16)(f.x * sc); h[1] = (_Float16)(f.y * sc);
  h[2] = (_Float16)(f.z * sc); h[3] = (_Float16)(f.w * sc);
  ((v4h*)outh)[i] = h;
}

// ---------------------------------------------------------------------------
// Edge scatter: one wave per edge, lane handles 4 features (f16 gather from L2,
// f32 atomic accumulate). Optional degree accumulation (layer 1 only).
// ---------------------------------------------------------------------------
__global__ __launch_bounds__(256) void scatter_f16_kernel(
    const _Float16* __restrict__ feat, const int* __restrict__ src,
    const int* __restrict__ dst, float* __restrict__ agg,
    float* __restrict__ deg, int n_edges) {
  const int t = blockIdx.x * blockDim.x + threadIdx.x;
  const int e = t >> 5;
  const int lane = t & 31;
  if (e >= n_edges) return;
  const int s = src[e];
  const int d = dst[e];
  v4h v = ((const v4h*)(feat + (size_t)s * D))[lane];
  float* ap = agg + (size_t)d * D + lane * 4;
#pragma unroll
  for (int i = 0; i < 4; ++i) atomicAdd(ap + i, (float)v[i]);
  if (deg != nullptr && lane == 0) atomicAdd(&deg[d], 1.0f);
}

// ---------------------------------------------------------------------------
// Layer 1: h1 = relu(mean @ W1l^T + b1l + x @ W1r^T)   (f16 out)
// Block = NT node tiles; weights staged to LDS once (TDM) and reused NT times.
// All 8 A tiles are batch-loaded into registers before the WMMA chain so the
// 16 b128 loads issue as clauses and drain with progressive loadcnt waits.
// ---------------------------------------------------------------------------
__global__ __launch_bounds__(256) void sage1_kernel(
    const _Float16* __restrict__ xh, const _Float16* __restrict__ meanh,
    const _Float16* __restrict__ W1l, const float* __restrict__ b1l,
    const _Float16* __restrict__ W1r, _Float16* __restrict__ h1h) {
  extern __shared__ char smem_raw[];
  _Float16* wl = (_Float16*)smem_raw;        // 128 x WPITCH
  _Float16* wr = wl + 128 * WPITCH;

  const int wave = threadIdx.x >> 5;
  const int lane = threadIdx.x & 31;
  const int half = lane >> 4;
  const int r = lane & 15;

#if HAVE_TDM
  if (wave == 0) {
    tdm_load_f16(lds_off(wl), W1l, 128);
    tdm_load_f16(lds_off(wr), W1r, 128);
    __builtin_amdgcn_s_wait_tensorcnt(0);
  }
#else
  stage_w_fallback(wl, W1l, 128);
  stage_w_fallback(wr, W1r, 128);
#endif
  __syncthreads();

  const int bcol = wave * 16 + r;
  const float bias = b1l[bcol];

  // Weight B tiles: LDS -> registers once, reused for all NT node tiles.
  v16h bl[4], br[4];
#pragma unroll
  for (int kt = 0; kt < 4; ++kt) {
    const int kb = kt * 32 + half * 8;
    bl[kt] = ldB_f16(wl + bcol * WPITCH, kb);
    br[kt] = ldB_f16(wr + bcol * WPITCH, kb);
  }

  for (int t = 0; t < NT; ++t) {
    const int tile = blockIdx.x * NT + t;
    const int arow = tile * 16 + r;
    const _Float16* mp = meanh + (size_t)arow * D;
    const _Float16* xp = xh + (size_t)arow * D;

    v16h a0[4], a1[4];
#pragma unroll
    for (int kt = 0; kt < 4; ++kt) {
      const int klo = kt * 32 + half * 8;
      a0[kt] = ldA_f16(mp, klo);
      a1[kt] = ldA_f16(xp, klo);
    }
    v8f c = {};
#pragma unroll
    for (int kt = 0; kt < 4; ++kt) {
      c = __builtin_amdgcn_wmma_f32_16x16x32_f16(false, a0[kt], false, bl[kt],
                                                 (short)0, c, false, false);
      c = __builtin_amdgcn_wmma_f32_16x16x32_f16(false, a1[kt], false, br[kt],
                                                 (short)0, c, false, false);
    }
#pragma unroll
    for (int i = 0; i < 8; ++i) {
      const int m = half * 8 + i;
      const float v = fmaxf(c[i] + bias, 0.0f);
      h1h[(size_t)(tile * 16 + m) * D + bcol] = (_Float16)v;
    }
  }
}

// ---------------------------------------------------------------------------
// Layer 2 + classifier:
//   emb = relu(mean2 @ W2l^T + b2l + h1 @ W2r^T) -> LDS (pitch WPITCH)
//   out = emb @ Wc^T + bc                         (waves 0..3)
// ---------------------------------------------------------------------------
__global__ __launch_bounds__(256) void sage2_cls_kernel(
    const _Float16* __restrict__ h1h, const _Float16* __restrict__ meanh,
    const _Float16* __restrict__ W2l, const float* __restrict__ b2l,
    const _Float16* __restrict__ W2r, const _Float16* __restrict__ Wc,
    const float* __restrict__ bc, float* __restrict__ out) {
  extern __shared__ char smem_raw[];
  _Float16* wl = (_Float16*)smem_raw;         // 128 x WPITCH
  _Float16* wr = wl + 128 * WPITCH;           // 128 x WPITCH
  _Float16* wc = wr + 128 * WPITCH;           // 64  x WPITCH
  _Float16* emb = wc + 64 * WPITCH;           // 16  x WPITCH

  const int wave = threadIdx.x >> 5;
  const int lane = threadIdx.x & 31;
  const int half = lane >> 4;
  const int r = lane & 15;

#if HAVE_TDM
  if (wave == 0) {
    tdm_load_f16(lds_off(wl), W2l, 128);
    tdm_load_f16(lds_off(wr), W2r, 128);
    tdm_load_f16(lds_off(wc), Wc, 64);
    __builtin_amdgcn_s_wait_tensorcnt(0);
  }
#else
  stage_w_fallback(wl, W2l, 128);
  stage_w_fallback(wr, W2r, 128);
  stage_w_fallback(wc, Wc, 64);
#endif
  __syncthreads();

  const int bcol = wave * 16 + r;
  const float bias = b2l[bcol];
  const int ccol = (wave & 3) * 16 + r;
  const _Float16* wcp = wc + ccol * WPITCH;

  // Weight B tiles: LDS -> registers once, reused for all NT node tiles.
  v16h bl[4], br[4];
#pragma unroll
  for (int kt = 0; kt < 4; ++kt) {
    const int kb = kt * 32 + half * 8;
    bl[kt] = ldB_f16(wl + bcol * WPITCH, kb);
    br[kt] = ldB_f16(wr + bcol * WPITCH, kb);
  }

  for (int t = 0; t < NT; ++t) {
    const int tile = blockIdx.x * NT + t;
    const int arow = tile * 16 + r;
    const _Float16* mp = meanh + (size_t)arow * D;
    const _Float16* hp = h1h + (size_t)arow * D;

    v16h a0[4], a1[4];
#pragma unroll
    for (int kt = 0; kt < 4; ++kt) {
      const int klo = kt * 32 + half * 8;
      a0[kt] = ldA_f16(mp, klo);
      a1[kt] = ldA_f16(hp, klo);
    }
    v8f c = {};
#pragma unroll
    for (int kt = 0; kt < 4; ++kt) {
      c = __builtin_amdgcn_wmma_f32_16x16x32_f16(false, a0[kt], false, bl[kt],
                                                 (short)0, c, false, false);
      c = __builtin_amdgcn_wmma_f32_16x16x32_f16(false, a1[kt], false, br[kt],
                                                 (short)0, c, false, false);
    }
    __syncthreads();  // previous iteration's classifier reads of emb are done
#pragma unroll
    for (int i = 0; i < 8; ++i) {
      const int m = half * 8 + i;
      emb[m * WPITCH + bcol] = (_Float16)fmaxf(c[i] + bias, 0.0f);
    }
    __syncthreads();  // emb visible to all waves

    if (wave < 4) {  // wave-uniform branch: EXEC all-ones inside, WMMA legal
      v8f c2 = {};
#pragma unroll
      for (int kt = 0; kt < 4; ++kt) {
        const int klo = kt * 32 + half * 8;
        const int kb = kt * 32 + half * 8;
        v16h a = ldA_f16(&emb[r * WPITCH], klo);  // ds_load path
        v16h b = ldB_f16(wcp, kb);
        c2 = __builtin_amdgcn_wmma_f32_16x16x32_f16(false, a, false, b,
                                                    (short)0, c2, false, false);
      }
      const float bias2 = bc[ccol];
#pragma unroll
      for (int i = 0; i < 8; ++i) {
        const int m = half * 8 + i;
        out[(size_t)(tile * 16 + m) * N_CLS + ccol] = c2[i] + bias2;
      }
    }
  }
}

// ---------------------------------------------------------------------------
extern "C" void kernel_launch(void* const* d_in, const int* in_sizes, int n_in,
                              void* d_out, int out_size, void* d_ws,
                              size_t ws_size, hipStream_t stream) {
  (void)in_sizes; (void)n_in; (void)out_size; (void)ws_size;

  const float* x   = (const float*)d_in[0];
  const int*   ei  = (const int*)d_in[1];   // edge_index [2, E]
  const float* W1l = (const float*)d_in[2];
  const float* b1l = (const float*)d_in[3];
  const float* W1r = (const float*)d_in[4];
  const float* W2l = (const float*)d_in[5];
  const float* b2l = (const float*)d_in[6];
  const float* W2r = (const float*)d_in[7];
  const float* Wc  = (const float*)d_in[8];
  const float* bc  = (const float*)d_in[9];
  float* out = (float*)d_out;

  const int* src = ei;
  const int* dst = ei + N_EDGES;

  // Workspace: deg | agg f32 | h1 f16 | x f16 | mean f16 | weights f16
  char* ws = (char*)d_ws;
  size_t off = 0;
  float* deg = (float*)(ws + off);
  off += (((size_t)N_NODES * 4) + 255) & ~(size_t)255;
  float* agg = (float*)(ws + off);
  off += (size_t)N_NODES * D * 4;
  _Float16* h1h = (_Float16*)(ws + off);
  off += (size_t)N_NODES * D * 2;
  _Float16* xh = (_Float16*)(ws + off);
  off += (size_t)N_NODES * D * 2;
  _Float16* meanh = (_Float16*)(ws + off);
  off += (size_t)N_NODES * D * 2;
  _Float16* w1lh = (_Float16*)(ws + off); off += (size_t)D * D * 2;
  _Float16* w1rh = (_Float16*)(ws + off); off += (size_t)D * D * 2;
  _Float16* w2lh = (_Float16*)(ws + off); off += (size_t)D * D * 2;
  _Float16* w2rh = (_Float16*)(ws + off); off += (size_t)D * D * 2;
  _Float16* wch  = (_Float16*)(ws + off); off += (size_t)N_CLS * D * 2;

  const int sblocks = (N_EDGES * 32 + 255) / 256;
  const int n4 = N_NODES * D / 4;
  const int nblk4 = (n4 + 255) / 256;
  const size_t lds1 = (size_t)2 * 128 * WPITCH * 2;                        // 69632 B
  const size_t lds2 = ((size_t)2 * 128 + 64 + 16) * WPITCH * 2;           // 91392 B

  // f16 pre-conversion
  cvt4_kernel<<<nblk4, 256, 0, stream>>>(x, xh, n4);
  cvt4_kernel<<<(D * D / 4 + 255) / 256, 256, 0, stream>>>(W1l, w1lh, D * D / 4);
  cvt4_kernel<<<(D * D / 4 + 255) / 256, 256, 0, stream>>>(W1r, w1rh, D * D / 4);
  cvt4_kernel<<<(D * D / 4 + 255) / 256, 256, 0, stream>>>(W2l, w2lh, D * D / 4);
  cvt4_kernel<<<(D * D / 4 + 255) / 256, 256, 0, stream>>>(W2r, w2rh, D * D / 4);
  cvt4_kernel<<<(N_CLS * D / 4 + 255) / 256, 256, 0, stream>>>(Wc, wch, N_CLS * D / 4);

  hipMemsetAsync(deg, 0, (size_t)N_NODES * sizeof(float), stream);
  hipMemsetAsync(agg, 0, (size_t)N_NODES * D * sizeof(float), stream);
  scatter_f16_kernel<<<sblocks, 256, 0, stream>>>(xh, src, dst, agg, deg, N_EDGES);
  mean_cvt_kernel<<<nblk4, 256, 0, stream>>>(agg, deg, meanh, n4);
  sage1_kernel<<<NBLK, 256, lds1, stream>>>(xh, meanh, w1lh, b1l, w1rh, h1h);
  hipMemsetAsync(agg, 0, (size_t)N_NODES * D * sizeof(float), stream);
  scatter_f16_kernel<<<sblocks, 256, 0, stream>>>(h1h, src, dst, agg, nullptr, N_EDGES);
  mean_cvt_kernel<<<nblk4, 256, 0, stream>>>(agg, deg, meanh, n4);
  sage2_cls_kernel<<<NBLK, 256, lds2, stream>>>(h1h, meanh, w2lh, b2l, w2rh,
                                                wch, bc, out);
}
